// GAT_41291815583870
// MI455X (gfx1250) — matrix operand order repaired and checked by reading, not compiled, not run on hardware
//
#include <hip/hip_runtime.h>
#include <hip/hip_bf16.h>

#define H_HEADS 4
#define DHEAD   32
#define HD      128
#define CCLS    10
#define NEG_SLOPE 0.2f

typedef __bf16 bf16x8  __attribute__((ext_vector_type(8)));
typedef __bf16 bf16x16 __attribute__((ext_vector_type(16)));
typedef float  f32x8   __attribute__((ext_vector_type(8)));

__device__ __forceinline__ float lrelu(float x) { return x > 0.f ? x : NEG_SLOPE * x; }

__device__ __forceinline__ void atomicMaxF(float* addr, float val) {
    unsigned int* ua = (unsigned int*)addr;
    unsigned int cur = *ua;
    while (__uint_as_float(cur) < val) {
        unsigned int assumed = cur;
        cur = atomicCAS(ua, assumed, __float_as_uint(val));
        if (cur == assumed) break;
    }
}

// ---------------- conversion kernels ----------------
__global__ void k_f32_to_bf16(const float* __restrict__ in, __bf16* __restrict__ out, int n) {
    int i = blockIdx.x * blockDim.x + threadIdx.x;
    if (i < n) out[i] = (__bf16)in[i];
}

// Wt[n*128+k] = bf16(W[k*128+n])  (transpose so B-fragment loads are contiguous)
__global__ void k_transpose_w_bf16(const float* __restrict__ W, __bf16* __restrict__ Wt) {
    int i = blockIdx.x * blockDim.x + threadIdx.x;
    if (i < HD * HD) {
        int n = i >> 7, k = i & 127;
        Wt[i] = (__bf16)W[k * HD + n];
    }
}

// ---------------- WMMA GEMM: C[Mpad x 128] = A[Mpad x 128] * W[128 x 128] ----------------
// A bf16 row-major (padded to 16-row multiple); Bt bf16 = W transposed; C f32 row-major (padded).
// One wave -> one 16x16 tile; block = 8 waves = one M-tile across all 8 N-tiles.
// Buffers are padded, so no bounds checks anywhere: straight-line loads/WMMA/stores.
__global__ void k_gemm_bf16_wmma(const __bf16* __restrict__ A, const __bf16* __restrict__ Bt,
                                 float* __restrict__ C) {
    const int lane  = threadIdx.x & 31;
    const int tileN = threadIdx.x >> 5;     // 0..7
    const int tileM = blockIdx.x;
    const int row   = lane & 15;
    const int half  = lane >> 4;

    const __bf16* arow = A  + (size_t)(tileM * 16 + row) * HD;
    const __bf16* bcol = Bt + (size_t)(tileN * 16 + row) * HD + half * 16;

    f32x8 c = {0.f, 0.f, 0.f, 0.f, 0.f, 0.f, 0.f, 0.f};
#pragma unroll
    for (int k0 = 0; k0 < HD; k0 += 32) {
        // A 16x32 bf16 fragment: lane holds row `row`, 16 consecutive K starting at k0+half*16
        bf16x8 a0 = *(const bf16x8*)(arow + k0 + half * 16);
        bf16x8 a1 = *(const bf16x8*)(arow + k0 + half * 16 + 8);
        // B 32x16 bf16 fragment: lane holds column `row`, 16 consecutive K starting at k0+half*16
        bf16x8 b0 = *(const bf16x8*)(bcol + k0);
        bf16x8 b1 = *(const bf16x8*)(bcol + k0 + 8);
        bf16x16 a, b;
#pragma unroll
        for (int i = 0; i < 8; ++i) { a[i] = a0[i]; a[i + 8] = a1[i]; b[i] = b0[i]; b[i + 8] = b1[i]; }
        c = __builtin_amdgcn_wmma_f32_16x16x32_bf16(false, a, false, b, (short)0, c, false, false);
    }
    // C layout: vgpr r -> (M = half*8 + r, N = row) within the tile
    float* crow = C + (size_t)(tileM * 16 + half * 8) * HD + tileN * 16 + row;
#pragma unroll
    for (int r = 0; r < 8; ++r) crow[(size_t)r * HD] = c[r];
}

// ---------------- per-node attention coefficients + state init ----------------
__global__ void k_node_alpha(const float* __restrict__ Hf, const float* __restrict__ a_src,
                             const float* __restrict__ a_dst, float* __restrict__ aS,
                             float* __restrict__ aD, float* __restrict__ mm,
                             float* __restrict__ dn, float* __restrict__ ACC, int N) {
    int t = blockIdx.x * blockDim.x + threadIdx.x;
    if (t >= N * H_HEADS) return;
    int n = t >> 2, h = t & 3;
    const float* hp = Hf + (size_t)n * HD + h * DHEAD;
    const float* as = a_src + h * DHEAD;
    const float* ad = a_dst + h * DHEAD;
    float* acc = ACC + (size_t)n * HD + h * DHEAD;
    float ss = 0.f, sd = 0.f;
#pragma unroll
    for (int d = 0; d < DHEAD; ++d) {
        float v = hp[d];
        ss += v * as[d];
        sd += v * ad[d];
        acc[d] = 0.f;
    }
    aS[t] = ss; aD[t] = sd;
    mm[t] = -__builtin_inff();
    dn[t] = 0.f;
}

// ---------------- edge pass 1: segment max over dst ----------------
__global__ void k_edge_max(const int* __restrict__ ei, const float* __restrict__ aS,
                           const float* __restrict__ aD, float* __restrict__ mm, int E) {
    int t = blockIdx.x * blockDim.x + threadIdx.x;
    if (t >= E * H_HEADS) return;
    int e = t >> 2, h = t & 3;
    int s = ei[e], d = ei[E + e];
    float v = lrelu(aS[s * H_HEADS + h] + aD[d * H_HEADS + h]);
    atomicMaxF(&mm[d * H_HEADS + h], v);
}

// ---------------- edge pass 2: fused weight + denom + weighted scatter-add ----------------
__global__ void k_edge_accum(const int* __restrict__ ei, const float* __restrict__ aS,
                             const float* __restrict__ aD, const float* __restrict__ mm,
                             const float* __restrict__ Hf, float* __restrict__ dn,
                             float* __restrict__ ACC, int E) {
    long long t = (long long)blockIdx.x * blockDim.x + threadIdx.x;
    if (t >= (long long)E * HD) return;
    int e = (int)(t >> 7);
    int d = (int)(t & 127);
    int h = d >> 5;
    int s = ei[e], dst = ei[E + e];
    float w = __expf(lrelu(aS[s * H_HEADS + h] + aD[dst * H_HEADS + h]) - mm[dst * H_HEADS + h]);
    if ((d & 31) == 0) atomicAdd(&dn[dst * H_HEADS + h], w);
    atomicAdd(&ACC[(size_t)dst * HD + d], w * Hf[(size_t)s * HD + d]);
}

// ---------------- node finalize: divide by denom, +bias, (relu), emit bf16 ----------------
__global__ void k_finalize(float* __restrict__ ACC, const float* __restrict__ dn,
                           const float* __restrict__ bias, __bf16* __restrict__ Abf,
                           int N, int apply_relu) {
    int t = blockIdx.x * blockDim.x + threadIdx.x;
    if (t >= N * HD) return;
    int n = t >> 7, d = t & 127;
    float v = ACC[t] / (dn[n * H_HEADS + (d >> 5)] + 1e-16f) + bias[d];
    if (apply_relu) v = fmaxf(v, 0.f);
    ACC[t] = v;
    Abf[t] = (__bf16)v;
}

// ---------------- pooling + classifier ----------------
__global__ void k_zero(float* __restrict__ p, int n) {
    int i = blockIdx.x * blockDim.x + threadIdx.x;
    if (i < n) p[i] = 0.f;
}

__global__ void k_pool(const float* __restrict__ Hfinal, const int* __restrict__ batch,
                       float* __restrict__ pooled, int N) {
    int t = blockIdx.x * blockDim.x + threadIdx.x;
    if (t >= N * HD) return;
    int n = t >> 7, d = t & 127;
    atomicAdd(&pooled[batch[n] * HD + d], Hfinal[t]);
}

__global__ void k_fc(const float* __restrict__ pooled, const float* __restrict__ Wfc,
                     const float* __restrict__ bfc, float* __restrict__ out, int G) {
    int t = blockIdx.x * blockDim.x + threadIdx.x;
    if (t >= G * CCLS) return;
    int g = t / CCLS, c = t % CCLS;
    float s = bfc[c];
    const float* p = pooled + (size_t)g * HD;
#pragma unroll
    for (int k = 0; k < HD; ++k) s += p[k] * Wfc[k * CCLS + c];
    out[t] = s;
}

extern "C" void kernel_launch(void* const* d_in, const int* in_sizes, int n_in,
                              void* d_out, int out_size, void* d_ws, size_t ws_size,
                              hipStream_t stream) {
    (void)n_in; (void)ws_size;
    const float* x     = (const float*)d_in[0];
    const int*   ei    = (const int*)d_in[1];
    const int*   batch = (const int*)d_in[2];
    const float* W[3]    = { (const float*)d_in[3], (const float*)d_in[7],  (const float*)d_in[11] };
    const float* asrc[3] = { (const float*)d_in[4], (const float*)d_in[8],  (const float*)d_in[12] };
    const float* adst[3] = { (const float*)d_in[5], (const float*)d_in[9],  (const float*)d_in[13] };
    const float* bias[3] = { (const float*)d_in[6], (const float*)d_in[10], (const float*)d_in[14] };
    const float* Wfc = (const float*)d_in[15];
    const float* bfc = (const float*)d_in[16];
    float* out = (float*)d_out;

    const int N    = in_sizes[0] / HD;          // F_IN == HD == 128
    const int Npad = (N + 15) & ~15;            // pad to 16-row multiple -> unguarded WMMA tiles
    const int E    = in_sizes[1] / 2;
    const int G    = out_size / CCLS;

    char* ws = (char*)d_ws;
    size_t off = 0;
    auto carve = [&](size_t bytes) -> char* {
        char* p = ws + off;
        off = (off + bytes + 255) & ~(size_t)255;
        return p;
    };
    __bf16* Abf    = (__bf16*)carve((size_t)Npad * HD * 2);   // bf16 activations (GEMM input, padded)
    float*  Hf     = (float*)carve((size_t)Npad * HD * 4);    // linear output h (padded)
    float*  ACC    = (float*)carve((size_t)N * HD * 4);       // attention-aggregated output
    float*  aS     = (float*)carve((size_t)N * H_HEADS * 4);
    float*  aD     = (float*)carve((size_t)N * H_HEADS * 4);
    float*  mm     = (float*)carve((size_t)N * H_HEADS * 4);
    float*  dn     = (float*)carve((size_t)N * H_HEADS * 4);
    __bf16* Wt     = (__bf16*)carve((size_t)HD * HD * 2);
    float*  pooled = (float*)carve((size_t)G * HD * 4);

    const int BLK = 256;
    k_f32_to_bf16<<<(N * HD + BLK - 1) / BLK, BLK, 0, stream>>>(x, Abf, N * HD);

    for (int L = 0; L < 3; ++L) {
        k_transpose_w_bf16<<<(HD * HD + BLK - 1) / BLK, BLK, 0, stream>>>(W[L], Wt);
        k_gemm_bf16_wmma<<<Npad / 16, BLK, 0, stream>>>(Abf, Wt, Hf);
        k_node_alpha<<<(N * H_HEADS + BLK - 1) / BLK, BLK, 0, stream>>>(
            Hf, asrc[L], adst[L], aS, aD, mm, dn, ACC, N);
        k_edge_max<<<(E * H_HEADS + BLK - 1) / BLK, BLK, 0, stream>>>(ei, aS, aD, mm, E);
        long long tot = (long long)E * HD;
        k_edge_accum<<<(int)((tot + BLK - 1) / BLK), BLK, 0, stream>>>(
            ei, aS, aD, mm, Hf, dn, ACC, E);
        k_finalize<<<(N * HD + BLK - 1) / BLK, BLK, 0, stream>>>(
            ACC, dn, bias[L], Abf, N, (L < 2) ? 1 : 0);
    }

    k_zero<<<(G * HD + BLK - 1) / BLK, BLK, 0, stream>>>(pooled, G * HD);
    k_pool<<<(N * HD + BLK - 1) / BLK, BLK, 0, stream>>>(ACC, batch, pooled, N);
    k_fc<<<(G * CCLS + BLK - 1) / BLK, BLK, 0, stream>>>(pooled, Wfc, bfc, out, G);
}
